// Head_11424613007808
// MI455X (gfx1250) — matrix operand order repaired and checked by reading, not compiled, not run on hardware
//
#include <hip/hip_runtime.h>

// ---------------------------------------------------------------------------
// Causal single-head attention for MI455X (gfx1250), wave32 + WMMA bf16.
//   out = softmax(mask(QK^T * C^-0.5)) @ V,  Q/K/V = x @ Wq/Wk/Wv
// Kernel 1: QKV projection (fp32 -> bf16), V stored TRANSPOSED [B][H][T].
// Kernel 2: one independent wave32 per 16 query rows, per-wave causal bound,
//           online softmax, v_wmma_f32_16x16x32_bf16 everywhere.
//           K/V^T key-block tiles are streamed into LDS by the Tensor Data
//           Mover (tensor_load_to_lds, TENSORcnt double-buffering) with
//           TDM-applied row padding for conflict-free ds_load_b128 fragments.
// ---------------------------------------------------------------------------

typedef __attribute__((ext_vector_type(16))) __bf16 v16bf;
typedef __attribute__((ext_vector_type(8)))  __bf16 v8bf;
typedef __attribute__((ext_vector_type(4)))  __bf16 v4bf;
typedef __attribute__((ext_vector_type(8)))  float  v8f;
typedef __attribute__((ext_vector_type(4)))  float  v4f;
typedef __attribute__((ext_vector_type(4)))  unsigned int u32x4;
typedef __attribute__((ext_vector_type(8)))  int          i32x8;
typedef __attribute__((ext_vector_type(4)))  int          i32x4;

#define B_ 4
#define T_ 2048
#define C_ 1024
#define H_ 128

// LDS row padding for manual tiles: 40 bf16 = 80 B.
#define LDSPAD 40

#if defined(__gfx1250__) && __has_builtin(__builtin_amdgcn_tensor_load_to_lds) \
                         && __has_builtin(__builtin_amdgcn_s_wait_tensorcnt)
#define USE_TDM 1
#else
#define USE_TDM 0
#endif

__device__ __forceinline__ v16bf cat8(v8bf lo, v8bf hi) {
  return __builtin_shufflevector(lo, hi, 0, 1, 2, 3, 4, 5, 6, 7,
                                         8, 9, 10, 11, 12, 13, 14, 15);
}

__device__ __forceinline__ v8f wmma_bf16(v16bf a, v16bf b, v8f c) {
  return __builtin_amdgcn_wmma_f32_16x16x32_bf16(false, a, false, b,
                                                 (short)0, c, false, false);
}

#if USE_TDM
// 2D TDM load of a (tile_h x tile_w) bf16 tile into LDS, with LDS row
// padding (pad_amount_code+1 dwords every 2^(pad_interval_code+1) dwords).
// Descriptor layout per CDNA5 ISA sec. 8 (D# groups 0/1; 2D -> rest = 0).
// This toolchain's builtin takes 6 args (clang-23 / therock-10.0 form):
//   (u32x4 g0, i32x8 g1, i32x4, i32x4, i32x8, i32 cpol)
__device__ __forceinline__ void tdm_load_2d_bf16(
    unsigned int lds_off, const void* gaddr,
    unsigned int tile_w, unsigned int tile_h,
    unsigned int tensor_w, unsigned int tensor_h, unsigned int row_stride,
    unsigned int pi_code, unsigned int pa_code)
{
  const unsigned long long ga = (unsigned long long)gaddr;
  u32x4 g0;
  g0[0] = 1u;                                        // count=1 (valid), user
  g0[1] = lds_off;                                   // lds_addr (bytes)
  g0[2] = (unsigned int)(ga & 0xFFFFFFFFu);          // global_addr[31:0]
  g0[3] = (unsigned int)((ga >> 32) & 0x01FFFFFFu)   // global_addr[56:32]
        | (2u << 30);                                // type = 2 ("image")
  i32x8 g1;
  g1[0] = (int)((1u << 16)                           // data_size = 2 bytes
              | (1u << 20)                           // pad_enable
              | (pi_code << 22) | (pa_code << 25));  // LDS padding
  g1[1] = (int)((tensor_w & 0xFFFFu) << 16);                      // dim0 lo
  g1[2] = (int)(((tensor_w >> 16) & 0xFFFFu)
              | ((tensor_h & 0xFFFFu) << 16));                    // dim0 hi|dim1 lo
  g1[3] = (int)(((tensor_h >> 16) & 0xFFFFu)
              | ((tile_w & 0xFFFFu) << 16));                      // dim1 hi|tile0
  g1[4] = (int)(tile_h & 0xFFFFu);                                // tile1, tile2=0
  g1[5] = (int)row_stride;                                        // dim0_stride lo
  g1[6] = 0;                                                      // stride hi
  g1[7] = 0;
  const i32x4 gz4 = {};
  const i32x8 gz8 = {};
  __builtin_amdgcn_tensor_load_to_lds(g0, g1, gz4, gz4, gz8, 0);
}
#endif

// ---------------------------------------------------------------------------
// Kernel 1: QKV projection.  grid = (B*T/128, 3), block = 256 (8 waves).
// blockIdx.y: 0 -> Q [B*T][H], 1 -> K [B*T][H], 2 -> V^T [B][H][T].
// ---------------------------------------------------------------------------
__global__ __launch_bounds__(256) void qkv_proj_kernel(
    const float* __restrict__ x,
    const float* __restrict__ Wq, const float* __restrict__ Wk,
    const float* __restrict__ Wv,
    __bf16* __restrict__ qo, __bf16* __restrict__ ko, __bf16* __restrict__ vo)
{
  __shared__ __align__(16) __bf16 Xs[128][LDSPAD];  // x tile,   [row][k]
  __shared__ __align__(16) __bf16 Wt[128][LDSPAD];  // W^T tile, [n]  [k]

  const int tid  = threadIdx.x;
  const int wave = tid >> 5;
  const int lane = tid & 31;
  const int half = lane >> 4;
  const int ln   = lane & 15;
  const int m0   = blockIdx.x * 128;

  const float* W;
  __bf16* out;
  if (blockIdx.y == 0)      { W = Wq; out = qo; }
  else if (blockIdx.y == 1) { W = Wk; out = ko; }
  else                      { W = Wv; out = vo; }

  v8f acc[8] = {};

  for (int k0 = 0; k0 < C_; k0 += 32) {
    for (int slot = tid; slot < 1024; slot += 256) {
      const int r = slot >> 3;
      const int c = (slot & 7) << 2;
      const v4f xv = *(const v4f*)&x[(size_t)(m0 + r) * C_ + (k0 + c)];
      v4bf bv;
      bv[0] = (__bf16)xv[0]; bv[1] = (__bf16)xv[1];
      bv[2] = (__bf16)xv[2]; bv[3] = (__bf16)xv[3];
      *(v4bf*)&Xs[r][c] = bv;
    }
    for (int slot = tid; slot < 1024; slot += 256) {
      const int kk = slot >> 5;
      const int n4 = (slot & 31) << 2;
      const v4f wv = *(const v4f*)&W[(size_t)(k0 + kk) * H_ + n4];
      Wt[n4 + 0][kk] = (__bf16)wv[0];
      Wt[n4 + 1][kk] = (__bf16)wv[1];
      Wt[n4 + 2][kk] = (__bf16)wv[2];
      Wt[n4 + 3][kk] = (__bf16)wv[3];
    }
    __syncthreads();

    const int arow = wave * 16 + ln;
    const int akb  = half * 16;
    const v16bf afrag = cat8(*(const v8bf*)&Xs[arow][akb],
                             *(const v8bf*)&Xs[arow][akb + 8]);
#pragma unroll
    for (int nt = 0; nt < 8; ++nt) {
      const __bf16* wp = &Wt[nt * 16 + ln][half * 16];
      const v16bf bfrag = cat8(*(const v8bf*)wp, *(const v8bf*)(wp + 8));
      acc[nt] = wmma_bf16(afrag, bfrag, acc[nt]);
    }
    __syncthreads();
  }

  if (blockIdx.y != 2) {
#pragma unroll
    for (int nt = 0; nt < 8; ++nt)
#pragma unroll
      for (int r = 0; r < 8; ++r)
        out[(size_t)(m0 + wave * 16 + r + 8 * half) * H_ + nt * 16 + ln] =
            (__bf16)acc[nt][r];
  } else {
    const int bb = m0 / T_;
    const int t0 = (m0 % T_) + wave * 16 + 8 * half;
    __bf16* vb = out + (size_t)bb * H_ * T_;
#pragma unroll
    for (int nt = 0; nt < 8; ++nt)
#pragma unroll
      for (int r = 0; r < 8; ++r)
        vb[(size_t)(nt * 16 + ln) * T_ + t0 + r] = (__bf16)acc[nt][r];
  }
}

// ---------------------------------------------------------------------------
// Kernel 2: causal flash attention.  grid = (T/16, B), block = 32 (1 wave).
// TDM double-buffers the K (32x128) and V^T (128x32) bf16 tiles in LDS.
// K LDS rows: 256B + 16B pad = 272B.  V^T LDS rows: 64B + 16B pad = 80B.
// ---------------------------------------------------------------------------
__global__ __launch_bounds__(32) void flash_attn_kernel(
    const __bf16* __restrict__ Q, const __bf16* __restrict__ K,
    const __bf16* __restrict__ Vt,   // [B][H][T]
    float* __restrict__ out)
{
  __shared__ __align__(16) __bf16 Ps[16][LDSPAD];  // wave-private P patch
#if USE_TDM
  __shared__ __align__(16) unsigned char Kls[2][32 * 272];
  __shared__ __align__(16) unsigned char Vls[2][128 * 80];
#endif

  const int lane = threadIdx.x & 31;
  const int half = lane >> 4;
  const int ln   = lane & 15;

  const int b    = blockIdx.y;
  const int qr0  = blockIdx.x * 16;
  const int base = b * T_;
  const __bf16* Vb = Vt + (size_t)b * H_ * T_;

  v16bf qfrag[4];
  {
    const __bf16* qp = Q + (size_t)(base + qr0 + ln) * H_ + half * 16;
#pragma unroll
    for (int hc4 = 0; hc4 < 4; ++hc4)
      qfrag[hc4] = cat8(*(const v8bf*)(qp + hc4 * 32),
                        *(const v8bf*)(qp + hc4 * 32 + 8));
  }

  v8f acc[8] = {};
  float mrow[8], lrow[8], alpha[8];
#pragma unroll
  for (int r = 0; r < 8; ++r) { mrow[r] = -3.0e38f; lrow[r] = 0.0f; }

  const float inv_scale = 0.03125f;               // C^-0.5 = 1/32
  const float LOG2E     = 1.4426950408889634f;
  const int   kend      = qr0 + 16;               // per-wave causal bound

#if USE_TDM
  const unsigned int koff0 = (unsigned int)(unsigned long long)&Kls[0][0];
  const unsigned int koff1 = (unsigned int)(unsigned long long)&Kls[1][0];
  const unsigned int voff0 = (unsigned int)(unsigned long long)&Vls[0][0];
  const unsigned int voff1 = (unsigned int)(unsigned long long)&Vls[1][0];
  // preload key block 0 into buffer 0
  tdm_load_2d_bf16(koff0, K + (size_t)base * H_, 128, 32, H_, T_, H_, 5, 3);
  tdm_load_2d_bf16(voff0, Vb, 32, 128, T_, H_, T_, 3, 3);
  int buf = 0;
#endif

  for (int kb = 0; kb < kend; kb += 32) {
#if USE_TDM
    if (kb + 32 < kend) {      // async-prefetch next block into other buffer
      tdm_load_2d_bf16(buf ? koff0 : koff1,
                       K + (size_t)(base + kb + 32) * H_, 128, 32,
                       H_, T_, H_, 5, 3);
      tdm_load_2d_bf16(buf ? voff0 : voff1,
                       Vb + kb + 32, 32, 128, T_, H_, T_, 3, 3);
      __builtin_amdgcn_s_wait_tensorcnt(2);  // current buffer's 2 TDMs done
    } else {
      __builtin_amdgcn_s_wait_tensorcnt(0);
    }
    __asm__ __volatile__("" ::: "memory");   // keep ds reads below the wait
    const unsigned char* kls = &Kls[buf][0];
    const unsigned char* vls = &Vls[buf][0];
#else
    if (kb + 32 < kend) {
      __builtin_prefetch(K + (size_t)(base + kb + 32 + lane) * H_, 0, 1);
#pragma unroll
      for (int j = 0; j < 4; ++j)
        __builtin_prefetch(Vb + (size_t)(lane * 4 + j) * T_ + kb + 32, 0, 1);
    }
#endif

    // S = Q @ K^T : two 16x16 tiles (key cols kb..+15, kb+16..+31)
    v8f s0 = {}, s1 = {};
    {
#if USE_TDM
      const unsigned char* kp0 = kls + (size_t)ln * 272 + half * 32;
      const unsigned char* kp1 = kls + (size_t)(16 + ln) * 272 + half * 32;
#pragma unroll
      for (int hc4 = 0; hc4 < 4; ++hc4) {
        const v16bf b0 = cat8(*(const v8bf*)(kp0 + hc4 * 64),
                              *(const v8bf*)(kp0 + hc4 * 64 + 16));
        s0 = wmma_bf16(qfrag[hc4], b0, s0);
        const v16bf b1 = cat8(*(const v8bf*)(kp1 + hc4 * 64),
                              *(const v8bf*)(kp1 + hc4 * 64 + 16));
        s1 = wmma_bf16(qfrag[hc4], b1, s1);
      }
#else
      const __bf16* kp0 = K + (size_t)(base + kb + ln) * H_ + half * 16;
      const __bf16* kp1 = kp0 + 16 * H_;
#pragma unroll
      for (int hc4 = 0; hc4 < 4; ++hc4) {
        const v16bf b0 = cat8(*(const v8bf*)(kp0 + hc4 * 32),
                              *(const v8bf*)(kp0 + hc4 * 32 + 8));
        s0 = wmma_bf16(qfrag[hc4], b0, s0);
        const v16bf b1 = cat8(*(const v8bf*)(kp1 + hc4 * 32),
                              *(const v8bf*)(kp1 + hc4 * 32 + 8));
        s1 = wmma_bf16(qfrag[hc4], b1, s1);
      }
#endif
    }

    // scale + causal mask + online softmax (16-lane row reductions, wave32)
#pragma unroll
    for (int r = 0; r < 8; ++r) {
      const int row = qr0 + r + 8 * half;
      const float a0 = (kb + ln      <= row) ? s0[r] * inv_scale : -3.0e38f;
      const float a1 = (kb + 16 + ln <= row) ? s1[r] * inv_scale : -3.0e38f;

      float mx = fmaxf(a0, a1);
#pragma unroll
      for (int off = 1; off < 16; off <<= 1)
        mx = fmaxf(mx, __shfl_xor(mx, off, 32));
      const float mnew = fmaxf(mrow[r], mx);

      alpha[r] = __builtin_amdgcn_exp2f((mrow[r] - mnew) * LOG2E);
      const float p0 = __builtin_amdgcn_exp2f((a0 - mnew) * LOG2E);
      const float p1 = __builtin_amdgcn_exp2f((a1 - mnew) * LOG2E);

      float rs = p0 + p1;
#pragma unroll
      for (int off = 1; off < 16; off <<= 1)
        rs += __shfl_xor(rs, off, 32);

      lrow[r] = lrow[r] * alpha[r] + rs;
      mrow[r] = mnew;

      Ps[r + 8 * half][ln]      = (__bf16)p0;   // same-wave DS: in order
      Ps[r + 8 * half][16 + ln] = (__bf16)p1;
    }

    // P A-fragment (16x32) back from LDS in A layout
    const v16bf pfrag = cat8(*(const v8bf*)&Ps[ln][half * 16],
                             *(const v8bf*)&Ps[ln][half * 16 + 8]);

    // O = O*alpha + P @ V
#pragma unroll
    for (int hc = 0; hc < 8; ++hc) {
#if USE_TDM
      const unsigned char* vp = vls + (size_t)(hc * 16 + ln) * 80 + half * 32;
      const v16bf vfrag = cat8(*(const v8bf*)vp, *(const v8bf*)(vp + 16));
#else
      const __bf16* vp = Vb + (size_t)(hc * 16 + ln) * T_ + kb + half * 16;
      const v16bf vfrag = cat8(*(const v8bf*)vp, *(const v8bf*)(vp + 8));
#endif
      v8f t = acc[hc];
#pragma unroll
      for (int r = 0; r < 8; ++r) t[r] *= alpha[r];
      acc[hc] = wmma_bf16(pfrag, vfrag, t);
    }
#if USE_TDM
    buf ^= 1;
#endif
  }

  float inv_l[8];
#pragma unroll
  for (int r = 0; r < 8; ++r) inv_l[r] = 1.0f / lrow[r];
#pragma unroll
  for (int hc = 0; hc < 8; ++hc)
#pragma unroll
    for (int r = 0; r < 8; ++r)
      out[(size_t)(base + qr0 + r + 8 * half) * H_ + hc * 16 + ln] =
          acc[hc][r] * inv_l[r];
}

// ---------------------------------------------------------------------------
extern "C" void kernel_launch(void* const* d_in, const int* in_sizes, int n_in,
                              void* d_out, int out_size, void* d_ws,
                              size_t ws_size, hipStream_t stream)
{
  (void)in_sizes; (void)n_in; (void)out_size; (void)ws_size;

  const float* x  = (const float*)d_in[0];
  const float* Wq = (const float*)d_in[1];
  const float* Wk = (const float*)d_in[2];
  const float* Wv = (const float*)d_in[3];
  // d_in[4] = causal mask (int32) — causality computed analytically, unused.

  const size_t BTH = (size_t)B_ * T_ * H_;
  __bf16* qw = (__bf16*)d_ws;        // Q   [B*T][H]   2 MB
  __bf16* kw = qw + BTH;             // K   [B*T][H]   2 MB
  __bf16* vw = kw + BTH;             // V^T [B][H][T]  2 MB  (6 MB of d_ws)

  dim3 gp((B_ * T_) / 128, 3);
  qkv_proj_kernel<<<gp, 256, 0, stream>>>(x, Wq, Wk, Wv, qw, kw, vw);

  dim3 ga(T_ / 16, B_);
  flash_attn_kernel<<<ga, 32, 0, stream>>>(qw, kw, vw, (float*)d_out);
}
